// Performer_43061342110437
// MI455X (gfx1250) — matrix operand order repaired and checked by reading, not compiled, not run on hardware
//
#include <hip/hip_runtime.h>

#define DEV __device__ __forceinline__

#ifndef __has_builtin
#define __has_builtin(x) 0
#endif

#if __has_builtin(__builtin_amdgcn_global_load_async_to_lds_b128)
#define HAS_ASYNC_LDS 1
#else
#define HAS_ASYNC_LDS 0
#endif

#if __has_builtin(__builtin_amdgcn_s_wait_asynccnt)
#define WAIT_ASYNC0() __builtin_amdgcn_s_wait_asynccnt(0)
#else
#define WAIT_ASYNC0() asm volatile("s_wait_asynccnt 0x0" ::: "memory")
#endif

typedef __bf16 bf16x16 __attribute__((ext_vector_type(16)));
typedef float  f32x8   __attribute__((ext_vector_type(8)));
typedef float  f32x4   __attribute__((ext_vector_type(4)));
typedef float  f32x2   __attribute__((ext_vector_type(2)));
typedef unsigned int u32x4 __attribute__((ext_vector_type(4)));
typedef int    i32x4   __attribute__((ext_vector_type(4)));

struct U32x8 { u32x4 lo, hi; };
union Pack8 { unsigned short u[8]; u32x4 v; };
union Pack4 { unsigned short u[4]; unsigned long long v; };

DEV unsigned short f2bf(float f) {
  unsigned u = __float_as_uint(f);
  u += 0x7fffu + ((u >> 16) & 1u);          // round-to-nearest-even
  return (unsigned short)(u >> 16);
}
DEV float bf2f(unsigned short h) { return __uint_as_float(((unsigned)h) << 16); }

// Assemble a v16bf WMMA operand from two 16-byte chunks (LDS or global).
DEV bf16x16 mk_frag(const unsigned short* p0, const unsigned short* p1) {
  U32x8 t;
  t.lo = *(const u32x4*)p0;
  t.hi = *(const u32x4*)p1;
  return __builtin_bit_cast(bf16x16, t);
}
DEV f32x8 wmma_bf16(bf16x16 a, bf16x16 b, f32x8 c) {
  return __builtin_amdgcn_wmma_f32_16x16x32_bf16(false, a, false, b, (short)0, c, false, false);
}

#if HAS_ASYNC_LDS
// 16B global -> LDS via the async memory path (ASYNCcnt tracked).
// Builtin signature wants pointers to 4xi32 vectors in AS1/AS3.
DEV void async_cp16(const unsigned short* g, unsigned short* l) {
  __builtin_amdgcn_global_load_async_to_lds_b128(
      (__attribute__((address_space(1))) i32x4*)(unsigned long long)g,
      (__attribute__((address_space(3))) i32x4*)(unsigned)(unsigned long long)l,
      0, 0);
}
#endif

static constexpr int   kS  = 2048, kB = 16, kH = 8, kD = 512;
static constexpr int   kN  = kS * kB;                 // 32768 token rows
static constexpr float kRATIO = 0.35355339059327373f; // 64^-0.25
static constexpr float kSTAB  = 1e-3f;

// ---------------------------------------------------------------------------
// Generic 128x128 tiled GEMM:  C[row][col] = A[row][:] . W[col][:] + bias
// A: bf16 [rows x lda] (K contiguous), W: bf16 [cols x ldw] (K contiguous)
// BK=64 double-buffered stages; async global->LDS when available.
// ---------------------------------------------------------------------------
enum { EB16 = 0, EB16RELU = 1, EB16T = 2, EF32RES = 3 };

template <int EPI>
__global__ __launch_bounds__(256) void perf_gemm_k(
    const unsigned short* __restrict__ A, int lda,
    const unsigned short* __restrict__ W, int ldw,
    const float* __restrict__ bias,
    const float* __restrict__ resid,
    unsigned short* __restrict__ outB,
    float* __restrict__ outF,
    long long ldo, int K)
{
  __shared__ alignas(16) unsigned short As[2][128 * 72];
  __shared__ alignas(16) unsigned short Bs[2][128 * 72];
  const int tid = threadIdx.x, wid = tid >> 5, lane = tid & 31;
  const int li = lane & 15, hi = lane >> 4;
  const int tm = blockIdx.y * 128, tn = blockIdx.x * 128;
  const int wm = (wid >> 2) * 64, wn = (wid & 3) * 32;
  const int lr = tid >> 1, lc = (tid & 1) * 32;
  const unsigned short* gA = A + (size_t)(tm + lr) * lda + lc;
  const unsigned short* gW = W + (size_t)(tn + lr) * ldw + lc;
  f32x8 acc[4][2] = {};

  auto stage = [&](int buf, int k0) {
    unsigned short* la = &As[buf][lr * 72 + lc];
    unsigned short* lb = &Bs[buf][lr * 72 + lc];
#if HAS_ASYNC_LDS
#pragma unroll
    for (int j = 0; j < 4; j++) {
      async_cp16(gA + k0 + j * 8, la + j * 8);
      async_cp16(gW + k0 + j * 8, lb + j * 8);
    }
#else
    const u32x4* sa = (const u32x4*)(gA + k0);
    const u32x4* sw = (const u32x4*)(gW + k0);
    u32x4* da = (u32x4*)la;
    u32x4* db = (u32x4*)lb;
#pragma unroll
    for (int j = 0; j < 4; j++) { da[j] = sa[j]; db[j] = sw[j]; }
#endif
  };

  stage(0, 0);
  int buf = 0;
  for (int k0 = 0; k0 < K; k0 += 64, buf ^= 1) {
#if HAS_ASYNC_LDS
    WAIT_ASYNC0();
#endif
    __syncthreads();
    if (k0 + 64 < K) stage(buf ^ 1, k0 + 64);   // prefetch next stage while computing
#pragma unroll
    for (int ks = 0; ks < 64; ks += 32) {
      bf16x16 af[4], bq[2];
#pragma unroll
      for (int m = 0; m < 4; m++) {
        const unsigned short* p = &As[buf][(wm + m * 16 + li) * 72 + ks];
        af[m] = mk_frag(p + hi * 8, p + 16 + hi * 8);     // A 16x32 lane layout
      }
#pragma unroll
      for (int n = 0; n < 2; n++) {
        const unsigned short* p = &Bs[buf][(wn + n * 16 + li) * 72 + ks];
        bq[n] = mk_frag(p + hi * 8, p + 16 + hi * 8);    // B 32x16 lane layout
      }
#pragma unroll
      for (int m = 0; m < 4; m++)
#pragma unroll
        for (int n = 0; n < 2; n++)
          acc[m][n] = wmma_bf16(af[m], bq[n], acc[m][n]);
    }
  }

#pragma unroll
  for (int m = 0; m < 4; m++) {
    const int rbase = tm + wm + m * 16 + 8 * hi;
#pragma unroll
    for (int n = 0; n < 2; n++) {
      const int col = tn + wn + n * 16 + li;
      const float bv = bias ? bias[col] : 0.0f;
      if (EPI == EB16T) {           // transposed: 8 contiguous rows per lane
        Pack8 p;
#pragma unroll
        for (int r = 0; r < 8; r++) p.u[r] = f2bf(acc[m][n][r] + bv);
        *(u32x4*)(outB + (size_t)col * ldo + rbase) = p.v;
      } else {
#pragma unroll
        for (int r = 0; r < 8; r++) {
          const size_t row = (size_t)(rbase + r);
          const float v = acc[m][n][r] + bv;
          if (EPI == EB16)          outB[row * ldo + col] = f2bf(v);
          else if (EPI == EB16RELU) outB[row * ldo + col] = f2bf(fmaxf(v, 0.0f));
          else                      outF[row * ldo + col] = v + resid[row * ldo + col];
        }
      }
    }
  }
}

// ---------------------------------------------------------------------------
// Per-head 128x64 K=64 kernel.
// MODE 0: qp = relu(ratio*q.w)+stab  (row-major)       grid (N/128, H)
// MODE 1: kp -> kp_t[h*64+m][n]     (transposed)       grid (N/128, H)
// MODE 2: num = qp.kvT / den        (row-major attn)   grid (S/128, B*H)
// ---------------------------------------------------------------------------
template <int MODE>
__global__ __launch_bounds__(256) void perf_featnum_k(
    const unsigned short* __restrict__ A,
    const unsigned short* __restrict__ Wall,
    unsigned short* __restrict__ out,
    const float* __restrict__ den,
    int Ntot)
{
  __shared__ alignas(16) unsigned short As[128 * 72];
  __shared__ alignas(16) unsigned short Ws[64 * 72];
  const int tid = threadIdx.x, wid = tid >> 5, lane = tid & 31;
  const int li = lane & 15, hi = lane >> 4;
  int h, rowbase;
  const unsigned short* W;
  if (MODE == 2) {
    const int pair = blockIdx.y;
    h = pair & 7;
    rowbase = (pair >> 3) * kS + blockIdx.x * 128;
    W = Wall + (size_t)pair * 4096;          // kvT[b][h] : [64 d][64 m]
  } else {
    h = blockIdx.y; rowbase = blockIdx.x * 128; W = Wall;
  }
  {
    const int r = tid >> 1, c = (tid & 1) * 32;
    const unsigned short* s = A + (size_t)(rowbase + r) * kD + h * 64 + c;
    unsigned short* d = &As[r * 72 + c];
    const int r2 = tid >> 2, c2 = (tid & 3) * 16;
    const unsigned short* s2 = W + r2 * 64 + c2;
    unsigned short* d2 = &Ws[r2 * 72 + c2];
#if HAS_ASYNC_LDS
#pragma unroll
    for (int j = 0; j < 4; j++) async_cp16(s + j * 8, d + j * 8);
#pragma unroll
    for (int j = 0; j < 2; j++) async_cp16(s2 + j * 8, d2 + j * 8);
    WAIT_ASYNC0();
#else
    const u32x4* sv = (const u32x4*)s;
    u32x4* dv = (u32x4*)d;
#pragma unroll
    for (int j = 0; j < 4; j++) dv[j] = sv[j];
    const u32x4* sv2 = (const u32x4*)s2;
    u32x4* dv2 = (u32x4*)d2;
    dv2[0] = sv2[0]; dv2[1] = sv2[1];
#endif
  }
  __syncthreads();

  f32x8 acc[4] = {};
  const int am = wid * 16;
#pragma unroll
  for (int ks = 0; ks < 64; ks += 32) {
    const unsigned short* ab = &As[(am + li) * 72 + ks];
    const bf16x16 a = mk_frag(ab + hi * 8, ab + 16 + hi * 8);
#pragma unroll
    for (int n = 0; n < 4; n++) {
      const unsigned short* bb = &Ws[(n * 16 + li) * 72 + ks];
      acc[n] = wmma_bf16(a, mk_frag(bb + hi * 8, bb + 16 + hi * 8), acc[n]);
    }
  }

  const int rbase = rowbase + am + 8 * hi;
#pragma unroll
  for (int n = 0; n < 4; n++) {
    const int col = n * 16 + li;
    if (MODE == 1) {
      Pack8 p;
#pragma unroll
      for (int r = 0; r < 8; r++)
        p.u[r] = f2bf(fmaxf(kRATIO * acc[n][r], 0.0f) + kSTAB);
      *(u32x4*)(out + (size_t)(h * 64 + col) * Ntot + rbase) = p.v;
    } else if (MODE == 0) {
#pragma unroll
      for (int r = 0; r < 8; r++)
        out[(size_t)(rbase + r) * kD + h * 64 + col] =
            f2bf(fmaxf(kRATIO * acc[n][r], 0.0f) + kSTAB);
    } else {
#pragma unroll
      for (int r = 0; r < 8; r++) {
        const float dn = den[(size_t)(rbase + r) * kH + h];
        out[(size_t)(rbase + r) * kD + h * 64 + col] = f2bf(acc[n][r] / dn);
      }
    }
  }
}

// ---------------------------------------------------------------------------
// kv[b,h] = kp^T @ v over S, via direct-from-global WMMA fragments.
// Output kvT[b][h][d][m] bf16. One block per (b,h); 8 waves K-split + LDS reduce.
// ---------------------------------------------------------------------------
__global__ __launch_bounds__(256) void perf_kv_k(
    const unsigned short* __restrict__ kpt, const unsigned short* __restrict__ vt,
    unsigned short* __restrict__ kvT, int Ntot)
{
  __shared__ float red[64 * 68];
  const int pair = blockIdx.x, b = pair >> 3, h = pair & 7;
  const int tid = threadIdx.x, wid = tid >> 5, lane = tid & 31;
  const int li = lane & 15, hi = lane >> 4;
  f32x8 acc[4][4] = {};
  const size_t colbase = (size_t)b * kS + wid * 256;
  const unsigned short* kb = kpt + (size_t)h * 64 * Ntot;
  const unsigned short* vb = vt + (size_t)h * 64 * Ntot;

  for (int ks = 0; ks < 256; ks += 32) {
    const size_t c = colbase + ks;
    bf16x16 af[4], bq[4];
#pragma unroll
    for (int m = 0; m < 4; m++) {
      const unsigned short* p = kb + (size_t)(m * 16 + li) * Ntot + c;
      af[m] = mk_frag(p + hi * 8, p + 16 + hi * 8);
    }
#pragma unroll
    for (int d = 0; d < 4; d++) {
      const unsigned short* p = vb + (size_t)(d * 16 + li) * Ntot + c;
      bq[d] = mk_frag(p + hi * 8, p + 16 + hi * 8);
    }
#pragma unroll
    for (int m = 0; m < 4; m++)
#pragma unroll
      for (int d = 0; d < 4; d++)
        acc[m][d] = wmma_bf16(af[m], bq[d], acc[m][d]);
  }
  // deterministic sequential cross-wave reduction
  for (int w = 0; w < 8; w++) {
    if (wid == w) {
#pragma unroll
      for (int m = 0; m < 4; m++)
#pragma unroll
        for (int d = 0; d < 4; d++)
#pragma unroll
          for (int r = 0; r < 8; r++) {
            const int Mr = m * 16 + r + 8 * hi, Nc = d * 16 + li;
            float* p = &red[Mr * 68 + Nc];
            if (w == 0) *p = acc[m][d][r]; else *p += acc[m][d][r];
          }
    }
    __syncthreads();
  }
  const int dcol = tid >> 2, mq = (tid & 3) * 16;
  Pack8 p0, p1;
#pragma unroll
  for (int j = 0; j < 8; j++) {
    p0.u[j] = f2bf(red[(mq + j) * 68 + dcol]);
    p1.u[j] = f2bf(red[(mq + 8 + j) * 68 + dcol]);
  }
  u32x4* dst = (u32x4*)(kvT + (size_t)pair * 4096 + dcol * 64 + mq);
  dst[0] = p0.v; dst[1] = p1.v;
}

// ksum[b][h][m] = sum_s kp_t[h*64+m][b*S+s]
__global__ __launch_bounds__(256) void perf_ksum_k(
    const unsigned short* __restrict__ kpt, float* __restrict__ ks, int Ntot)
{
  const int wid = threadIdx.x >> 5, lane = threadIdx.x & 31;
  const int r = blockIdx.x * 8 + wid;
  const int b = r >> 9, hm = r & 511;
  const unsigned short* p = kpt + (size_t)hm * Ntot + (size_t)b * kS;
  float s = 0.f;
#pragma unroll
  for (int i = 0; i < 8; i++) {
    u32x4 q = *(const u32x4*)(p + i * 256 + lane * 8);
#pragma unroll
    for (int j = 0; j < 4; j++) {
      s += bf2f((unsigned short)(q[j] & 0xffffu)) + bf2f((unsigned short)(q[j] >> 16));
    }
  }
#pragma unroll
  for (int off = 16; off > 0; off >>= 1) s += __shfl_xor(s, off, 32);
  if (lane == 0) ks[r] = s;
}

// den[n][h] = qp[n,h,:].ksum[b,h,:]
__global__ __launch_bounds__(256) void perf_den_k(
    const unsigned short* __restrict__ qp, const float* __restrict__ ks,
    float* __restrict__ den)
{
  const int idx = blockIdx.x * 256 + threadIdx.x;
  const int n = idx >> 3, h = idx & 7, b = n >> 11;
  const unsigned short* qr = qp + (size_t)n * kD + h * 64;
  const float* kr = ks + b * 512 + h * 64;
  float s = 0.f;
#pragma unroll
  for (int m = 0; m < 64; m++) s += bf2f(qr[m]) * kr[m];
  den[idx] = s;
}

// LayerNorm over 512 cols; writes fp32 residual stream + bf16 copy.
__global__ __launch_bounds__(256) void perf_ln_k(
    const float* __restrict__ y, const float* __restrict__ sc,
    const float* __restrict__ bi, float* __restrict__ xo,
    unsigned short* __restrict__ xb)
{
  const int wid = threadIdx.x >> 5, lane = threadIdx.x & 31;
  const size_t row = (size_t)blockIdx.x * 8 + wid;
  const float* yr = y + row * kD;
  f32x4 v[4];
  float s = 0.f, s2 = 0.f;
#pragma unroll
  for (int i = 0; i < 4; i++) {
    v[i] = *(const f32x4*)(yr + i * 128 + lane * 4);
#pragma unroll
    for (int j = 0; j < 4; j++) { s += v[i][j]; s2 += v[i][j] * v[i][j]; }
  }
#pragma unroll
  for (int off = 16; off > 0; off >>= 1) {
    s += __shfl_xor(s, off, 32); s2 += __shfl_xor(s2, off, 32);
  }
  const float mean = s * (1.0f / 512.0f);
  const float var = s2 * (1.0f / 512.0f) - mean * mean;
  const float rstd = rsqrtf(var + 1e-5f);
#pragma unroll
  for (int i = 0; i < 4; i++) {
    const int c0 = i * 128 + lane * 4;
    f32x4 o; Pack4 pk;
#pragma unroll
    for (int j = 0; j < 4; j++) {
      o[j] = (v[i][j] - mean) * rstd * sc[c0 + j] + bi[c0 + j];
      pk.u[j] = f2bf(o[j]);
    }
    *(f32x4*)(xo + row * kD + c0) = o;
    *(unsigned long long*)(xb + row * kD + c0) = pk.v;
  }
}

__global__ __launch_bounds__(256) void perf_prep_k(
    const float* __restrict__ x, const float* __restrict__ pos,
    unsigned short* __restrict__ qk, unsigned short* __restrict__ xb)
{
  const size_t e = ((size_t)blockIdx.x * 256 + threadIdx.x) * 4;
  const f32x4 xv = *(const f32x4*)(x + e);
  const f32x4 pv = *(const f32x4*)(pos + e);
  Pack4 a, b;
#pragma unroll
  for (int j = 0; j < 4; j++) { a.u[j] = f2bf(xv[j] + pv[j]); b.u[j] = f2bf(xv[j]); }
  *(unsigned long long*)(qk + e) = a.v;
  *(unsigned long long*)(xb + e) = b.v;
}

__global__ __launch_bounds__(256) void perf_conv_k(
    const float* __restrict__ s, unsigned short* __restrict__ d)
{
  const size_t e = ((size_t)blockIdx.x * 256 + threadIdx.x) * 4;
  const f32x4 v = *(const f32x4*)(s + e);
  Pack4 p;
#pragma unroll
  for (int j = 0; j < 4; j++) p.u[j] = f2bf(v[j]);
  *(unsigned long long*)(d + e) = p.v;
}

__global__ __launch_bounds__(256) void perf_copy_k(
    const float* __restrict__ s, float* __restrict__ d)
{
  const size_t e = ((size_t)blockIdx.x * 256 + threadIdx.x) * 4;
  *(f32x4*)(d + e) = *(const f32x4*)(s + e);
}

// [B,S,D] internal -> [S,B,D] output
__global__ __launch_bounds__(256) void perf_perm_k(
    const float* __restrict__ x, float* __restrict__ o)
{
  const int n = blockIdx.x;
  const int b = n >> 11, sidx = n & 2047;
  const size_t orow = (size_t)sidx * kB + b;
  const int c = threadIdx.x * 2;
  *(f32x2*)(o + orow * kD + c) = *(const f32x2*)(x + (size_t)n * kD + c);
}

// ---------------------------------------------------------------------------
extern "C" void kernel_launch(void* const* d_in, const int* in_sizes, int n_in,
                              void* d_out, int out_size, void* d_ws, size_t ws_size,
                              hipStream_t stream)
{
  (void)in_sizes; (void)n_in; (void)out_size; (void)ws_size;
  const float* src  = (const float*)d_in[0];
  const float* pos  = (const float*)d_in[2];
  const float* inw  = (const float*)d_in[3];
  const float* inb  = (const float*)d_in[4];
  const float* oww  = (const float*)d_in[5];
  const float* obb  = (const float*)d_in[6];
  const float* w1   = (const float*)d_in[7];
  const float* b1   = (const float*)d_in[8];
  const float* w2   = (const float*)d_in[9];
  const float* b2   = (const float*)d_in[10];
  const float* n1s  = (const float*)d_in[11];
  const float* n1b  = (const float*)d_in[12];
  const float* n2s  = (const float*)d_in[13];
  const float* n2b  = (const float*)d_in[14];
  const float* worf = (const float*)d_in[15];
  float* out = (float*)d_out;

  char* ws = (char*)d_ws;
  size_t off = 0;
  auto alloc = [&](size_t bytes) { size_t o = off; off += (bytes + 255) & ~(size_t)255; return o; };
  const size_t N = kN, D = kD;
  float* xf = (float*)(ws + alloc(N * D * 4));
  float* yf = (float*)(ws + alloc(N * D * 4));
  unsigned short* qk  = (unsigned short*)(ws + alloc(N * D * 2)); // also attn_bf
  unsigned short* xb  = (unsigned short*)(ws + alloc(N * D * 2));
  unsigned short* qb  = (unsigned short*)(ws + alloc(N * D * 2)); // f1 alias base
  unsigned short* kbq = (unsigned short*)(ws + alloc(N * D * 2));
  unsigned short* qp  = (unsigned short*)(ws + alloc(N * D * 2));
  unsigned short* kpt = (unsigned short*)(ws + alloc(N * D * 2));
  unsigned short* vt  = (unsigned short*)(ws + alloc(N * D * 2));
  unsigned short* f1  = qb;                                       // 128 MB alias (qb..kpt dead by FFN time)
  unsigned short* kvt  = (unsigned short*)(ws + alloc(128 * 4096 * 2));
  float* ksumB = (float*)(ws + alloc(16 * 512 * 4));
  float* denB  = (float*)(ws + alloc(N * 8 * 4));
  unsigned short* wqkv = (unsigned short*)(ws + alloc(1536 * 512 * 2));
  unsigned short* wo   = (unsigned short*)(ws + alloc(512 * 512 * 2));
  unsigned short* wl1  = (unsigned short*)(ws + alloc(2048 * 512 * 2));
  unsigned short* wl2  = (unsigned short*)(ws + alloc(512 * 2048 * 2));
  unsigned short* wf   = (unsigned short*)(ws + alloc(64 * 64 * 2));

  const dim3 T(256);
  perf_copy_k<<<16384, T, 0, stream>>>(src, xf);  // x = src ([B,S,D] internal)

  for (int i = 0; i < 6; i++) {
    perf_conv_k<<<768,  T, 0, stream>>>(inw  + (size_t)i * 1536 * 512, wqkv);
    perf_conv_k<<<256,  T, 0, stream>>>(oww  + (size_t)i * 512 * 512,  wo);
    perf_conv_k<<<1024, T, 0, stream>>>(w1   + (size_t)i * 2048 * 512, wl1);
    perf_conv_k<<<1024, T, 0, stream>>>(w2   + (size_t)i * 512 * 2048, wl2);
    perf_conv_k<<<4,    T, 0, stream>>>(worf + (size_t)i * 4096,       wf);

    perf_prep_k<<<16384, T, 0, stream>>>(xf, pos, qk, xb);

    // Q, K from qk; V from x (transposed epilogue -> v_t[h*64+d][n])
    perf_gemm_k<EB16><<<dim3(4, 256), T, 0, stream>>>(
        qk, 512, wqkv, 512, inb + (size_t)i * 1536, nullptr, qb, nullptr, 512, 512);
    perf_gemm_k<EB16><<<dim3(4, 256), T, 0, stream>>>(
        qk, 512, wqkv + 512 * 512, 512, inb + (size_t)i * 1536 + 512, nullptr, kbq, nullptr, 512, 512);
    perf_gemm_k<EB16T><<<dim3(4, 256), T, 0, stream>>>(
        xb, 512, wqkv + 1024 * 512, 512, inb + (size_t)i * 1536 + 1024, nullptr, vt, nullptr,
        (long long)N, 512);

    // performer feature maps
    perf_featnum_k<0><<<dim3(256, 8), T, 0, stream>>>(qb, wf, qp, nullptr, (int)N);
    perf_featnum_k<1><<<dim3(256, 8), T, 0, stream>>>(kbq, wf, kpt, nullptr, (int)N);

    perf_ksum_k<<<1024, T, 0, stream>>>(kpt, ksumB, (int)N);
    perf_kv_k<<<128, T, 0, stream>>>(kpt, vt, kvt, (int)N);
    perf_den_k<<<1024, T, 0, stream>>>(qp, ksumB, denB);

    // num/den -> attn (reuses qk buffer)
    perf_featnum_k<2><<<dim3(16, 128), T, 0, stream>>>(qp, kvt, qk, denB, (int)N);

    // out-proj + residual, LN1
    perf_gemm_k<EF32RES><<<dim3(4, 256), T, 0, stream>>>(
        qk, 512, wo, 512, obb + (size_t)i * 512, xf, nullptr, yf, 512, 512);
    perf_ln_k<<<4096, T, 0, stream>>>(yf, n1s + (size_t)i * 512, n1b + (size_t)i * 512, xf, xb);

    // FFN
    perf_gemm_k<EB16RELU><<<dim3(16, 256), T, 0, stream>>>(
        xb, 512, wl1, 512, b1 + (size_t)i * 2048, nullptr, f1, nullptr, 2048, 512);
    perf_gemm_k<EF32RES><<<dim3(4, 256), T, 0, stream>>>(
        f1, 2048, wl2, 2048, b2 + (size_t)i * 512, xf, nullptr, yf, 512, 2048);
    perf_ln_k<<<4096, T, 0, stream>>>(yf, n2s + (size_t)i * 512, n2b + (size_t)i * 512, xf, xb);
  }

  perf_perm_k<<<32768, T, 0, stream>>>(xf, out);
}
